// CausalAttentionForcing_30700426232360
// MI455X (gfx1250) — compile-verified, matrix operand order checked by reading, not run on hardware
//
#include <hip/hip_runtime.h>

// CausalAttentionForcing for MI455X (gfx1250), fused flash-attention style.
// B=32, S=1024, D=256.  All matmuls bf16 WMMA (v_wmma_f32_16x16x32_bf16),
// f32 accumulate.  causal_bias cancels inside softmax (uniform logit shift).
//
// Workspace layout (64 MB + 512 KB):
//   [0   , 16MB) : Qbf  bf16 [B*S, D]
//   [16MB, 32MB) : Kbf  bf16 [B*S, D]
//   [32MB, 48MB) : embI bf16 [B, D, S]  (t pair-interleaved per 32-block)
//   [48MB, 64MB) : Xbf  bf16 [B*S, D]   (emb pre-converted, row-major)
//   [64MB        +128KB) : Wqbf bf16 [D, D]
//   [64MB+128KB, +256KB) : Wkbf bf16 [D, D]
//   [64MB+256KB, +384KB) : doorw  f32 [B*S]  = w * (state==4|state==5)
//   [64MB+384KB, +512KB) : switchm f32 [B*S] = (state==3)

#define BB 32
#define SS 1024
#define DD 256

typedef __attribute__((ext_vector_type(16))) __bf16 v16bf;
typedef __attribute__((ext_vector_type(8)))  float  v8f;
typedef __attribute__((ext_vector_type(4)))  int    v4i_t;

// ---- CDNA5 async global->LDS path (guarded; fallback = load + ds_store) ----
#if defined(__has_builtin)
#if __has_builtin(__builtin_amdgcn_global_load_async_to_lds_b128)
#define HAVE_ASYNC_LDS 1
#endif
#if __has_builtin(__builtin_amdgcn_s_wait_asynccnt)
#define HAVE_WAIT_ASYNC 1
#endif
#endif
#ifndef HAVE_ASYNC_LDS
#define HAVE_ASYNC_LDS 0
#endif
#ifndef HAVE_WAIT_ASYNC
#define HAVE_WAIT_ASYNC 0
#endif

typedef __attribute__((address_space(1))) v4i_t gv4i_t;   // global int4
typedef __attribute__((address_space(3))) v4i_t lv4i_t;   // LDS int4

__device__ inline void stage16(const unsigned short* g, unsigned short* l) {
#if HAVE_ASYNC_LDS
  __builtin_amdgcn_global_load_async_to_lds_b128((gv4i_t*)g, (lv4i_t*)l,
                                                 0, 0);
#else
  *(uint4*)l = *(const uint4*)g;
#endif
}
__device__ inline void stage_wait_barrier() {
#if HAVE_ASYNC_LDS
#if HAVE_WAIT_ASYNC
  __builtin_amdgcn_s_wait_asynccnt(0);
#else
  asm volatile("s_wait_asynccnt 0" ::: "memory");
#endif
#endif
  __syncthreads();
}

// Pack two f32 -> two bf16 in one dword: bias round then v_perm_b32 grabs the
// two high halves (3 VALU total, no 16-bit register moves).
__device__ inline unsigned pack_bf2(float a, float b) {
  unsigned au = __float_as_uint(a) + 0x8000u;
  unsigned bu = __float_as_uint(b) + 0x8000u;
  return __builtin_amdgcn_perm(bu, au, 0x07060302u);
}
__device__ inline unsigned short f2bf_bits(float x) {
  return (unsigned short)((__float_as_uint(x) + 0x8000u) >> 16);
}

// A/B fragment for 16x16x32 bf16 WMMA from row-major bf16 memory (global or
// LDS) where the contraction dim (K) is contiguous.  ISA 7.12.2: lanes 0-15
// hold K {0..7}U{16..23}, lanes 16-31 hold K {8..15}U{24..31}; two contiguous
// 16-byte loads per lane.
__device__ inline v16bf load_frag_bf(const unsigned short* p, int lane) {
  int off = (lane & 16) ? 8 : 0;
  union { uint4 q[2]; v16bf v; } u;
  u.q[0] = *(const uint4*)(p + off);
  u.q[1] = *(const uint4*)(p + off + 16);
  return u.v;
}

__device__ inline v8f wmma_bf16(v16bf a, v16bf b, v8f c) {
  return __builtin_amdgcn_wmma_f32_16x16x32_bf16(false, a, false, b,
                                                 (short)0, c, false, false);
}

// ---------------------------------------------------------------------------
// Elementwise fp32 -> bf16 (4 elements/thread, float4 in, uint2 out).
__global__ void cvt_bf16_kernel(const float* __restrict__ src,
                                unsigned short* __restrict__ dst, int n4) {
  int i = blockIdx.x * blockDim.x + threadIdx.x;
  if (i >= n4) return;
  float4 v = ((const float4*)src)[i];
  uint2 o;
  o.x = pack_bf2(v.x, v.y);
  o.y = pack_bf2(v.z, v.w);
  ((uint2*)dst)[i] = o;
}

// ---------------------------------------------------------------------------
// Precompute per-token mask coefficients.
__global__ void mask_kernel(const long long* __restrict__ state,
                            const float* __restrict__ wptr,
                            float* __restrict__ doorw,
                            float* __restrict__ switchm) {
  int i = blockIdx.x * blockDim.x + threadIdx.x;
  if (i >= BB * SS) return;
  int st = (int)state[i];
  float w = wptr[0];
  doorw[i]   = (st == 4 || st == 5) ? w : 0.0f;
  switchm[i] = (st == 3) ? 1.0f : 0.0f;
}

// ---------------------------------------------------------------------------
// emb [B,S,D] f32 -> embI [B,D,S] bf16 with t pair-interleave
// [0,16,1,17,...,15,31] inside each 32-block.
__global__ __launch_bounds__(256) void embT_kernel(
    const float* __restrict__ emb, unsigned short* __restrict__ embI) {
  __shared__ float tile[32][65];
  int b = blockIdx.z, tblk = blockIdx.y, dblk = blockIdx.x;
  const float* src = emb + ((size_t)b * SS + tblk * 32) * DD + dblk * 64;
  int tid = threadIdx.x;
#pragma unroll
  for (int k = 0; k < 8; ++k) {
    int lin = tid + k * 256;
    int t = lin >> 6, d = lin & 63;
    tile[t][d] = src[(size_t)t * DD + d];
  }
  __syncthreads();
  int lane = tid & 31, wv = tid >> 5;
  int posl = 2 * (lane & 15) + (lane >> 4);
#pragma unroll
  for (int j = 0; j < 8; ++j) {
    int d = wv * 8 + j;
    embI[((size_t)b * DD + dblk * 64 + d) * SS + tblk * 32 + posl] =
        f2bf_bits(tile[lane][d]);
  }
}

// ---------------------------------------------------------------------------
// Q = emb*Wq^T + bq, K = emb*Wk^T + bk, bf16 in / bf16 out, f32 accum.
__global__ __launch_bounds__(256) void qk_proj_kernel(
    const unsigned short* __restrict__ Xbf,
    const unsigned short* __restrict__ Wqbf, const float* __restrict__ bq,
    const unsigned short* __restrict__ Wkbf, const float* __restrict__ bk,
    unsigned short* __restrict__ Qbf, unsigned short* __restrict__ Kbf) {
  int wave = blockIdx.x * (blockDim.x >> 5) + (threadIdx.x >> 5);
  int lane = threadIdx.x & 31;
  int colgrp  = wave & 3;
  int rowtile = (wave >> 2) & 2047;
  int qk      = wave >> 13;
  const unsigned short* W = qk ? Wkbf : Wqbf;
  const float* bias       = qk ? bk : bq;
  unsigned short* Y       = qk ? Kbf : Qbf;

  int la = lane & 15, hi = lane >> 4;
  int e0 = colgrp * 64;
  const unsigned short* arow = Xbf + (size_t)(rowtile * 16 + la) * DD;

  v8f zero8 = {0.f, 0.f, 0.f, 0.f, 0.f, 0.f, 0.f, 0.f};
  v8f acc[4] = {zero8, zero8, zero8, zero8};
#pragma unroll
  for (int kk = 0; kk < DD; kk += 32) {
    v16bf af = load_frag_bf(arow + kk, lane);
#pragma unroll
    for (int j = 0; j < 4; ++j) {
      int e = e0 + j * 16 + la;
      v16bf bf = load_frag_bf(W + (size_t)e * DD + kk, lane);
      acc[j] = wmma_bf16(af, bf, acc[j]);
    }
  }
#pragma unroll
  for (int j = 0; j < 4; ++j) {
    int e = e0 + j * 16 + la;
    float be = bias[e];
#pragma unroll
    for (int r = 0; r < 8; ++r) {
      int srow = rowtile * 16 + r + 8 * hi;
      Y[(size_t)srow * DD + e] = f2bf_bits(acc[j][r] + be);
    }
  }
}

// ---------------------------------------------------------------------------
// Fused flash attention.  One block = 8 waves = 8 consecutive 16-row s-tiles
// of the SAME batch; the per-step K-tile (32x256) and embI-tile (256x32) are
// staged ONCE per block into LDS (async global->LDS when available), cutting
// L2 read traffic 8x.  Tiles padded +8 bf16/row for LDS bank behavior.
#define KPAD 264   // 256 + 8 bf16
#define EPAD 40    // 32 + 8 bf16
#define NSTEP (SS / 32)

__global__ __launch_bounds__(256) void attn_kernel(
    const float* __restrict__ emb,
    const unsigned short* __restrict__ Qbf,
    const unsigned short* __restrict__ Kbf,
    const unsigned short* __restrict__ embI,
    const float* __restrict__ doorw,
    const float* __restrict__ switchm,
    float* __restrict__ out) {
  __shared__ unsigned short Kt[32 * KPAD];    // 16.5 KB
  __shared__ unsigned short Et[256 * EPAD];   // 20.0 KB
  __shared__ unsigned int   Pp[8][256];       //  8.0 KB
  int tid = threadIdx.x;
  int lane = tid & 31, wv = tid >> 5;
  int la = lane & 15, hi = lane >> 4;
  int b  = blockIdx.x >> 3;                       // 8 blocks per batch
  int s0 = (((blockIdx.x & 7) << 3) + wv) * 16;   // this wave's s-tile

  // Preload Q A-fragments for all 8 d-chunks (rows s0..s0+15).
  const unsigned short* Qrow = Qbf + ((size_t)b * SS + s0 + la) * DD;
  v16bf qf[8];
#pragma unroll
  for (int c = 0; c < 8; ++c) qf[c] = load_frag_bf(Qrow + c * 32, lane);

  float sw[8], m[8], l[8];
#pragma unroll
  for (int r = 0; r < 8; ++r) {
    sw[r] = switchm[b * SS + s0 + r + 8 * hi];
    m[r] = -1e30f;
    l[r] = 0.0f;
  }
  v8f zero8 = {0.f, 0.f, 0.f, 0.f, 0.f, 0.f, 0.f, 0.f};
  v8f acc[16];
#pragma unroll
  for (int j = 0; j < 16; ++j) acc[j] = zero8;

  unsigned int* lds = &Pp[wv][0];

  for (int step = 0; step < NSTEP; ++step) {
    int t0 = step * 32;
    // --- cooperative staging: K tile (32 rows x 256) + embI tile (256 x 32) -
    {
      const unsigned short* Ksrc = Kbf + ((size_t)b * SS + t0) * DD;
#pragma unroll
      for (int k = 0; k < 2; ++k) {            // 512 chunks of 8 bf16
        int id = tid + k * 256;
        int row = id >> 4, c = id & 15;
        stage16(Ksrc + row * DD + c * 8, Kt + row * KPAD + c * 8);
      }
      const unsigned short* Esrc = embI + (size_t)b * DD * SS + t0;
#pragma unroll
      for (int k = 0; k < 4; ++k) {            // 1024 chunks of 8 bf16
        int id = tid + k * 256;
        int d = id >> 2, c = id & 3;
        stage16(Esrc + (size_t)d * SS + c * 8, Et + d * EPAD + c * 8);
      }
    }
    stage_wait_barrier();

    // --- scores: 16x32 tile = two N-tiles, 8 K-chunks each (from LDS) ---
    v8f sc0 = zero8, sc1 = zero8;
#pragma unroll
    for (int c = 0; c < 8; ++c) {
      sc0 = wmma_bf16(qf[c], load_frag_bf(Kt + la * KPAD + c * 32, lane), sc0);
      sc1 = wmma_bf16(qf[c], load_frag_bf(Kt + (16 + la) * KPAD + c * 32, lane),
                      sc1);
    }
    // --- forced logits + online softmax ---
    float d0 = doorw[b * SS + t0 + la];
    float d1 = doorw[b * SS + t0 + 16 + la];
    float alpha[8];
#pragma unroll
    for (int r = 0; r < 8; ++r) {
      float f0 = sw[r] * d0 * sc0[r];
      float f1 = sw[r] * d1 * sc1[r];
      float mx = fmaxf(f0, f1);
      mx = fmaxf(mx, __shfl_xor(mx, 1, 32));
      mx = fmaxf(mx, __shfl_xor(mx, 2, 32));
      mx = fmaxf(mx, __shfl_xor(mx, 4, 32));
      mx = fmaxf(mx, __shfl_xor(mx, 8, 32));   // stays within 16-lane half
      float mn = fmaxf(m[r], mx);
      float a  = __expf(m[r] - mn);
      float p0 = __expf(f0 - mn), p1 = __expf(f1 - mn);
      float rs = p0 + p1;
      rs += __shfl_xor(rs, 1, 32);
      rs += __shfl_xor(rs, 2, 32);
      rs += __shfl_xor(rs, 4, 32);
      rs += __shfl_xor(rs, 8, 32);
      l[r] = l[r] * a + rs;
      m[r] = mn;
      alpha[r] = a;
      lds[(r + 8 * hi) * 16 + la] = pack_bf2(p0, p1);   // pair (t, t+16)
    }
    // rescale accumulators (VALU co-executes with the WMMAs below)
#pragma unroll
    for (int j = 0; j < 16; ++j)
#pragma unroll
      for (int r = 0; r < 8; ++r) acc[j][r] *= alpha[r];

    // --- P as A-fragment: 8 contiguous LDS dwords per lane ---
    union { uint4 q[2]; v16bf v; } af;
    const uint4* lptr = (const uint4*)(lds + la * 16 + hi * 8);
    af.q[0] = lptr[0];
    af.q[1] = lptr[1];

    // --- O += P * emb : B-fragments from staged pair-interleaved tile ---
#pragma unroll
    for (int j = 0; j < 16; ++j) {
      const unsigned short* bp = Et + (j * 16 + la) * EPAD + hi * 16;
      union { uint4 q[2]; v16bf v; } bfr;
      bfr.q[0] = ((const uint4*)bp)[0];
      bfr.q[1] = ((const uint4*)bp)[1];
      acc[j] = wmma_bf16(af.v, bfr.v, acc[j]);
    }
    __syncthreads();   // all waves done reading Kt/Et before next staging
  }

  // --- epilogue: out = emb + 0.5 * O / l ---
#pragma unroll
  for (int r = 0; r < 8; ++r) {
    int srow = s0 + r + 8 * hi;
    float inv = 0.5f / l[r];
    const float* er = emb + ((size_t)b * SS + srow) * DD;
    float* orow = out + ((size_t)b * SS + srow) * DD;
#pragma unroll
    for (int j = 0; j < 16; ++j) {
      int d = j * 16 + la;
      orow[d] = er[d] + acc[j][r] * inv;
    }
  }
}

// ---------------------------------------------------------------------------
extern "C" void kernel_launch(void* const* d_in, const int* in_sizes, int n_in,
                              void* d_out, int out_size, void* d_ws,
                              size_t ws_size, hipStream_t stream) {
  (void)in_sizes; (void)n_in; (void)out_size; (void)ws_size;
  const float*     emb   = (const float*)d_in[0];
  const long long* state = (const long long*)d_in[1];
  const float*     Wq    = (const float*)d_in[2];
  const float*     bq    = (const float*)d_in[3];
  const float*     Wk    = (const float*)d_in[4];
  const float*     bk    = (const float*)d_in[5];
  const float*     wgt   = (const float*)d_in[6];
  // d_in[7] (causal_bias) is a uniform logit shift -> cancels in softmax.

  const size_t MB = 1024 * 1024;
  char* ws = (char*)d_ws;
  unsigned short* Qbf  = (unsigned short*)(ws);
  unsigned short* Kbf  = (unsigned short*)(ws + 16 * MB);
  unsigned short* embI = (unsigned short*)(ws + 32 * MB);
  unsigned short* Xbf  = (unsigned short*)(ws + 48 * MB);
  unsigned short* Wqbf = (unsigned short*)(ws + 64 * MB);
  unsigned short* Wkbf = (unsigned short*)(ws + 64 * MB + 128 * 1024);
  float* doorw   = (float*)(ws + 64 * MB + 256 * 1024);
  float* switchm = (float*)(ws + 64 * MB + 384 * 1024);
  float* out = (float*)d_out;

  int n4emb = BB * SS * DD / 4;
  int n4w   = DD * DD / 4;
  hipLaunchKernelGGL(cvt_bf16_kernel, dim3(n4emb / 256), dim3(256), 0, stream,
                     emb, Xbf, n4emb);
  hipLaunchKernelGGL(cvt_bf16_kernel, dim3(n4w / 256), dim3(256), 0, stream,
                     Wq, Wqbf, n4w);
  hipLaunchKernelGGL(cvt_bf16_kernel, dim3(n4w / 256), dim3(256), 0, stream,
                     Wk, Wkbf, n4w);
  hipLaunchKernelGGL(mask_kernel, dim3(BB * SS / 256), dim3(256), 0, stream,
                     state, wgt, doorw, switchm);
  hipLaunchKernelGGL(embT_kernel, dim3(4, 32, 32), dim3(256), 0, stream,
                     emb, embI);
  hipLaunchKernelGGL(qk_proj_kernel, dim3(2048), dim3(256), 0, stream,
                     Xbf, Wqbf, bq, Wkbf, bk, Qbf, Kbf);
  hipLaunchKernelGGL(attn_kernel, dim3(256), dim3(256), 0, stream,
                     emb, Qbf, Kbf, embI, doorw, switchm, out);
}
